// AdaAT_31379031064960
// MI455X (gfx1250) — matrix operand order repaired and checked by reading, not compiled, not run on hardware
//
#include <hip/hip_runtime.h>
#include <hip/hip_bf16.h>

typedef __attribute__((ext_vector_type(16))) _Float16 v16h;
typedef __attribute__((ext_vector_type(8)))  _Float16 v8h;
typedef __attribute__((ext_vector_type(8)))  float    v8f;
typedef __attribute__((ext_vector_type(4)))  float    v4f;

#define BATCH 4096
#define PDIM  256
#define FCH   20
#define HH    18
#define WW    18
#define VOLN  (FCH * HH * WW)   // 6480
#define NOUT2 80                // 20 (scale) + 20 (angle) + 40 (trans)
#define PI_F  3.14159f

// ---------------------------------------------------------------------------
// para_code f32 -> f16, converted exactly once (GEMM1 A-operand)
// ---------------------------------------------------------------------------
__global__ __launch_bounds__(256) void cvt_pc_kernel(const float* __restrict__ pc,
                                                     _Float16* __restrict__ pc16) {
    int i = blockIdx.x * 256 + threadIdx.x;        // 4096*256 = 1048576 threads
    pc16[i] = (_Float16)pc[i];
}

// W1 (256x256 row-major, k-major contract dim) -> W1T f16, W1T[n*256 + k]
__global__ __launch_bounds__(256) void cvt_w1_kernel(const float* __restrict__ w1,
                                                     _Float16* __restrict__ w1t) {
    int i = blockIdx.x * 256 + threadIdx.x;        // 65536 threads, reads coalesced
    int n = i & 255;
    int k = i >> 8;
    w1t[n * 256 + k] = (_Float16)w1[k * 256 + n];
}

// Concatenate Ws(256x20) | Wr(256x20) | Wt(256x40) -> W2T[n*256 + k], n in [0,80)
__global__ __launch_bounds__(256) void cvt_w2_kernel(const float* __restrict__ ws,
                                                     const float* __restrict__ wr,
                                                     const float* __restrict__ wt,
                                                     _Float16* __restrict__ w2t) {
    int i = blockIdx.x * 256 + threadIdx.x;        // 80*256 = 20480 threads
    int n = i / 256;
    int k = i & 255;
    float v;
    if (n < 20)       v = ws[k * 20 + n];
    else if (n < 40)  v = wr[k * 20 + (n - 20)];
    else              v = wt[k * 40 + (n - 40)];
    w2t[i] = (_Float16)v;
}

// ---------------------------------------------------------------------------
// Shared WMMA microkernel body: one 16x16 tile per wave, f16 A/B, f32 acc.
// A (16x32 f16): lane holds row M=lane&15; elems 0-7 <-> K=16*half+0..7,
//                elems 8-15 <-> K=16*half+8..15  (two 16B loads)
// B (32x16 f16): lane holds col N=lane&15; elems 0-15 <-> K=16*half+e (32B load)
// C/D: lane holds col N=lane&15; VGPR i <-> row M=8*half+i
// ---------------------------------------------------------------------------
__device__ __forceinline__ v8f wmma_k256(const _Float16* __restrict__ aRow,
                                         const _Float16* __restrict__ bCol,
                                         int half) {
    v8f acc = {};
#pragma unroll
    for (int k0 = 0; k0 < 256; k0 += 32) {
        v8h a_lo = *(const v8h*)(aRow + k0 + half * 16);
        v8h a_hi = *(const v8h*)(aRow + k0 + half * 16 + 8);
        v16h A;
#pragma unroll
        for (int i = 0; i < 8; ++i) { A[i] = a_lo[i]; A[8 + i] = a_hi[i]; }
        v16h B = *(const v16h*)(bCol + k0 + half * 16);
        acc = __builtin_amdgcn_wmma_f32_16x16x32_f16(false, A, false, B,
                                                     (short)0, acc, false, false);
    }
    return acc;
}

// ---------------------------------------------------------------------------
// GEMM1: h16 = relu(pc16(4096x256) @ W1 + b1) -> f16 (4096x256)
// ---------------------------------------------------------------------------
__global__ __launch_bounds__(256) void gemm1_kernel(const _Float16* __restrict__ pc16,
                                                    const _Float16* __restrict__ w1t,
                                                    const float* __restrict__ b1,
                                                    _Float16* __restrict__ h16) {
    const int wave = threadIdx.x >> 5;
    const int lane = threadIdx.x & 31;
    const int tile = blockIdx.x * 8 + wave;        // 512 blocks * 8 = 4096 tiles
    const int nT = tile & 15;                      // 16 N-tiles (256/16)
    const int mT = tile >> 4;                      // 256 M-tiles (4096/16)
    const int r    = lane & 15;
    const int half = lane >> 4;

    v8f acc = wmma_k256(pc16 + (mT * 16 + r) * 256, w1t + (nT * 16 + r) * 256, half);

    const int  bcol = nT * 16 + r;
    const float bias = b1[bcol];
#pragma unroll
    for (int i = 0; i < 8; ++i) {
        int m = mT * 16 + half * 8 + i;
        float v = acc[i] + bias;
        v = v > 0.f ? v : 0.f;
        h16[m * 256 + bcol] = (_Float16)v;
    }
}

// ---------------------------------------------------------------------------
// GEMM2: out2 = h16(4096x256) @ [Ws|Wr|Wt](256x80) + bias -> f32 raw
// ---------------------------------------------------------------------------
__global__ __launch_bounds__(256) void gemm2_kernel(const _Float16* __restrict__ h16,
                                                    const _Float16* __restrict__ w2t,
                                                    const float* __restrict__ bs,
                                                    const float* __restrict__ br,
                                                    const float* __restrict__ bt,
                                                    float* __restrict__ out2) {
    const int wave = threadIdx.x >> 5;
    const int lane = threadIdx.x & 31;
    const int tile = blockIdx.x * 8 + wave;        // 160 blocks * 8 = 1280 tiles
    const int nT = tile % 5;                       // 5 N-tiles (80/16)
    const int mT = tile / 5;                       // 256 M-tiles
    const int r    = lane & 15;
    const int half = lane >> 4;

    v8f acc = wmma_k256(h16 + (mT * 16 + r) * 256, w2t + (nT * 16 + r) * 256, half);

    const int bcol = nT * 16 + r;
    float bias;
    if (bcol < 20)      bias = bs[bcol];
    else if (bcol < 40) bias = br[bcol - 20];
    else                bias = bt[bcol - 40];

#pragma unroll
    for (int i = 0; i < 8; ++i) {
        int m = mT * 16 + half * 8 + i;
        out2[m * NOUT2 + bcol] = acc[i] + bias;
    }
}

// ---------------------------------------------------------------------------
// Per-(b,f) affine coefficients: a11 a12 a21 a22 tx ty (stride 8 floats)
// ---------------------------------------------------------------------------
__global__ __launch_bounds__(256) void affine_kernel(const float* __restrict__ out2,
                                                     float* __restrict__ aff) {
    int i = blockIdx.x * 256 + threadIdx.x;        // 4096*20 = 81920 threads
    int b = i / FCH;
    int f = i % FCH;
    const float* o = out2 + b * NOUT2;
    float sc  = 2.f / (1.f + expf(-o[f]));         // 2 * sigmoid
    float ang = tanhf(o[20 + f]) * PI_F;
    float tx  = tanhf(o[40 + 2 * f]);
    float ty  = tanhf(o[40 + 2 * f + 1]);
    float c = cosf(ang);
    float s = sinf(ang);
    float* a = aff + i * 8;
    a[0] =  c * sc;   // a11
    a[1] = -s * sc;   // a12
    a[2] =  s * sc;   // a21
    a[3] =  c * sc;   // a22
    a[4] =  tx;
    a[5] =  ty;
}

// ---------------------------------------------------------------------------
// Trilinear grid-sample (align_corners=False, zero padding).
// One block per batch image: stage the 25.9KB volume + 640B affine table in
// LDS (320KB/WGP), then all 8 taps per output hit LDS instead of VMEM.
// HBM traffic = one sequential read of feature_map + one write of out.
// ---------------------------------------------------------------------------
__global__ __launch_bounds__(256) void sample_kernel(const float* __restrict__ fm,
                                                     const float* __restrict__ aff,
                                                     float* __restrict__ out) {
    __shared__ float vol[VOLN];          // 6480 floats = 25.9 KB
    __shared__ float affs[FCH * 8];      // 160 floats

    const int b   = blockIdx.x;          // 4096 blocks
    const int tid = threadIdx.x;

    const float* src = fm + (size_t)b * VOLN;
    for (int i = tid; i < VOLN; i += 256)
        vol[i] = src[i];
    if (tid < FCH * 8)
        affs[tid] = aff[(b * FCH) * 8 + tid];
    __syncthreads();

    float* dst = out + (size_t)b * VOLN;
    for (int idx = tid; idx < VOLN; idx += 256) {
        int x = idx % WW;
        int t = idx / WW;
        int y = t % HH;
        int f = t / HH;

        const float* a = affs + f * 8;
        float X = 2.f * (float)x / (float)(WW - 1) - 1.f;
        float Y = 2.f * (float)y / (float)(HH - 1) - 1.f;
        float gx = a[0] * X + a[1] * Y + a[4];
        float gy = a[2] * X + a[3] * Y + a[5];
        float gz = 2.f * (float)f / (float)(FCH - 1) - 1.f;

        float ix = ((gx + 1.f) * (float)WW - 1.f) * 0.5f;
        float iy = ((gy + 1.f) * (float)HH - 1.f) * 0.5f;
        float iz = ((gz + 1.f) * (float)FCH - 1.f) * 0.5f;

        float fx0 = floorf(ix), fy0 = floorf(iy), fz0 = floorf(iz);
        float wx = ix - fx0, wy = iy - fy0, wz = iz - fz0;
        int x0 = (int)fx0, y0 = (int)fy0, z0 = (int)fz0;

        float acc = 0.f;
#pragma unroll
        for (int cz = 0; cz < 2; ++cz) {
            float wzc = cz ? wz : 1.f - wz;
            int zi = z0 + cz;
#pragma unroll
            for (int cy = 0; cy < 2; ++cy) {
                float wyc = cy ? wy : 1.f - wy;
                int yi = y0 + cy;
#pragma unroll
                for (int cx = 0; cx < 2; ++cx) {
                    float wxc = cx ? wx : 1.f - wx;
                    int xi = x0 + cx;
                    if (xi >= 0 && xi < WW && yi >= 0 && yi < HH && zi >= 0 && zi < FCH)
                        acc += wzc * wyc * wxc * vol[(zi * HH + yi) * WW + xi];
                }
            }
        }
        dst[idx] = acc;
    }
}

// ---------------------------------------------------------------------------
extern "C" void kernel_launch(void* const* d_in, const int* in_sizes, int n_in,
                              void* d_out, int out_size, void* d_ws, size_t ws_size,
                              hipStream_t stream) {
    const float* fm  = (const float*)d_in[0];   // (4096,20,18,18)
    const float* pc  = (const float*)d_in[1];   // (4096,256)
    const float* W1  = (const float*)d_in[2];   // (256,256)
    const float* b1  = (const float*)d_in[3];   // (256,)
    const float* Ws  = (const float*)d_in[4];   // (256,20)
    const float* bs  = (const float*)d_in[5];   // (20,)
    const float* Wr  = (const float*)d_in[6];   // (256,20)
    const float* br  = (const float*)d_in[7];   // (20,)
    const float* Wt  = (const float*)d_in[8];   // (256,40)
    const float* bt  = (const float*)d_in[9];   // (40,)
    float* out = (float*)d_out;

    char* ws = (char*)d_ws;
    // scratch layout (all offsets 32B-aligned)
    _Float16* pc16 = (_Float16*)(ws);                                   // 2 MB
    _Float16* h16  = (_Float16*)(ws + 2097152);                         // 2 MB
    _Float16* w1t  = (_Float16*)(ws + 4194304);                         // 128 KB
    _Float16* w2t  = (_Float16*)(ws + 4194304 + 131072);                // 40 KB
    float*    out2 = (float*)   (ws + 4194304 + 131072 + 40960);        // 1.25 MB
    float*    aff  = (float*)   (ws + 4194304 + 131072 + 40960 + 1310720); // 2.5 MB

    cvt_pc_kernel<<<4096, 256, 0, stream>>>(pc, pc16);
    cvt_w1_kernel<<<256, 256, 0, stream>>>(W1, w1t);
    cvt_w2_kernel<<<80, 256, 0, stream>>>(Ws, Wr, Wt, w2t);
    gemm1_kernel<<<512, 256, 0, stream>>>(pc16, w1t, b1, h16);
    gemm2_kernel<<<160, 256, 0, stream>>>(h16, w2t, bs, br, bt, out2);
    affine_kernel<<<320, 256, 0, stream>>>(out2, aff);
    sample_kernel<<<4096, 256, 0, stream>>>(fm, aff, out);
}
